// TuneableAttentionMHA_2199023255964
// MI455X (gfx1250) — compile-verified, matrix operand order checked by reading, not run on hardware
//
#include <hip/hip_runtime.h>
#include <hip/hip_bf16.h>

// ---------------------------------------------------------------------------
// Types for CDNA5 WMMA
// ---------------------------------------------------------------------------
typedef __attribute__((ext_vector_type(16))) __bf16 v16bf;
typedef __attribute__((ext_vector_type(8)))  float  v8f;
typedef __attribute__((__vector_size__(16))) int    v4i;   // matches builtin param

union Frag16 {            // one 16x32 bf16 A/B fragment slice per lane (32 bytes)
    uint4 u[2];
    v16bf v;
};

__device__ __forceinline__ unsigned short f2bf(float f) {
    unsigned u = __float_as_uint(f);
    unsigned r = (u + 0x7FFFu + ((u >> 16) & 1u)) >> 16;
    return (unsigned short)r;
}
__device__ __forceinline__ unsigned pack2bf(float a, float b) {
    return (unsigned)f2bf(a) | ((unsigned)f2bf(b) << 16);
}
__device__ __forceinline__ uint4 pack8bf(float4 f0, float4 f1) {
    uint4 r;
    r.x = pack2bf(f0.x, f0.y);
    r.y = pack2bf(f0.z, f0.w);
    r.z = pack2bf(f1.x, f1.y);
    r.w = pack2bf(f1.z, f1.w);
    return r;
}
__device__ __forceinline__ uint4 pack8f(const float* p) {
    uint4 r;
    r.x = pack2bf(p[0], p[1]);
    r.y = pack2bf(p[2], p[3]);
    r.z = pack2bf(p[4], p[5]);
    r.w = pack2bf(p[6], p[7]);
    return r;
}

#define HN   16
#define RR   32
#define DKK  128
#define DD   2048
#define NTOK 2048
#define BB   4

// ---------------------------------------------------------------------------
// CDNA5 async global->LDS copy (ASYNCcnt-tracked), guarded so compile never
// fails if the toolchain doesn't declare the builtins.
// ---------------------------------------------------------------------------
#if defined(__gfx1250__) && __has_builtin(__builtin_amdgcn_global_load_async_to_lds_b128)
#define USE_ASYNC_LDS 1
#else
#define USE_ASYNC_LDS 0
#endif

#if USE_ASYNC_LDS
__device__ __forceinline__ __attribute__((address_space(1))) v4i* to_glb_v4i(const void* p) {
    return (__attribute__((address_space(1))) v4i*)(unsigned long long)p;
}
__device__ __forceinline__ __attribute__((address_space(3))) v4i* to_lds_v4i(void* p) {
    // generic LDS pointer low 32 bits == LDS byte offset (ISA 10.2)
    return (__attribute__((address_space(3))) v4i*)(unsigned int)(unsigned long long)p;
}
#endif

__device__ __forceinline__ void wait_async_lds() {
#if USE_ASYNC_LDS
#if __has_builtin(__builtin_amdgcn_s_wait_asynccnt)
    __builtin_amdgcn_s_wait_asynccnt(0);
#else
    asm volatile("s_wait_asynccnt 0" ::: "memory");
#endif
#endif
}

// Stage one 32-byte row chunk (16 bf16 elems) of a tile into LDS.
template<bool BF16SRC>
__device__ __forceinline__ void stage_chunk(const void* src, size_t gElem,
                                            unsigned short* ldsDst)
{
    if (BF16SRC) {
#if USE_ASYNC_LDS
        const unsigned short* g = (const unsigned short*)src + gElem;
        __builtin_amdgcn_global_load_async_to_lds_b128(to_glb_v4i(g), to_lds_v4i(ldsDst), 0, 0);
        __builtin_amdgcn_global_load_async_to_lds_b128(to_glb_v4i(g), to_lds_v4i(ldsDst), 16, 0);
#else
        const uint4* p = (const uint4*)((const unsigned short*)src + gElem);
        uint4* d = (uint4*)ldsDst;
        d[0] = p[0];
        d[1] = p[1];
#endif
    } else {
        const float4* p = (const float4*)((const float*)src + gElem);
        __builtin_prefetch((const float*)src + gElem + 64, 0, 1);   // 2 tiles ahead
        float4 f0 = p[0], f1 = p[1], f2 = p[2], f3 = p[3];
        uint4* d = (uint4*)ldsDst;
        d[0] = pack8bf(f0, f1);
        d[1] = pack8bf(f2, f3);
    }
}

// ---------------------------------------------------------------------------
// Kernel 1: effective low-rank weights, transposed for contiguous B-tile loads
//   WqUt[c][k] = sum_d Wq[h*128+d, k] * U[d, r]   with c = h*32 + r
// ---------------------------------------------------------------------------
__global__ __launch_bounds__(256) void effw_kernel(
    const float* __restrict__ Wq, const float* __restrict__ Wk,
    const float* __restrict__ U,
    unsigned short* __restrict__ WqUt, unsigned short* __restrict__ WkUt)
{
    int kcol = blockIdx.x * 256 + threadIdx.x;           // 0..2047
    int c    = blockIdx.y;                               // 0..511
    int h = c >> 5, r = c & 31;
    const float* W = blockIdx.z ? Wk : Wq;
    unsigned short* O = blockIdx.z ? WkUt : WqUt;
    const float* wcol = W + (size_t)(h * DKK) * DD + kcol;
    float acc = 0.f;
#pragma unroll 4
    for (int d = 0; d < DKK; ++d)
        acc += wcol[(size_t)d * DD] * U[d * RR + r];
    O[(size_t)c * DD + kcol] = f2bf(acc);
}

// ---------------------------------------------------------------------------
// Kernel 2: specialized LDS-tiled bf16 WMMA GEMM:  Out = A[MxK] * Bt[NxK]^T
//   Double-buffered K pipeline; bf16 sources staged via async LDS copies.
//   ABF/BBF: source element type is bf16 (else fp32, converted in-regs)
//   MODE 0: fp32 Out[M x Nout] row-major (times scale)
//   MODE 1: bf16 q/k permute:  Out[((b*16+h)*NTOK+n)*32 + r], c = h*32+r
//   MODE 2: bf16 V transpose:  Out[((b*16+h)*128+d)*NTOK + n], c = h*128+d
// ---------------------------------------------------------------------------
template<bool ABF, bool BBF, int MODE>
__global__ __launch_bounds__(256) void gemm_wmma_kernel(
    const void* __restrict__ Aptr, const void* __restrict__ Bptr,
    int M, int Nout, int K,
    void* __restrict__ Out, float scale)
{
    __shared__ __align__(16) unsigned short sA[2][128 * 32];
    __shared__ __align__(16) unsigned short sB[2][128 * 32];

    const int tid  = threadIdx.x;
    const int lane = tid & 31;
    const int wv   = tid >> 5;        // wave 0..7
    const int wm   = wv & 3;          // M sub-tile group (32 rows each)
    const int wn   = wv >> 2;         // N sub-tile group (64 cols each)
    const int half = lane >> 4;
    const int l16  = lane & 15;

    const int blockM = blockIdx.y * 128;
    const int blockN = blockIdx.x * 128;

    v8f acc[2][4];
#pragma unroll
    for (int i = 0; i < 2; ++i)
#pragma unroll
        for (int j = 0; j < 4; ++j)
#pragma unroll
            for (int e = 0; e < 8; ++e) acc[i][j][e] = 0.f;

    const int rowL = tid >> 1;        // 0..127 tile row loaded by this thread
    const int seg  = tid & 1;         // which 16-element half of the 32-wide k chunk
    const size_t gA0 = (size_t)(blockM + rowL) * K + seg * 16;
    const size_t gB0 = (size_t)(blockN + rowL) * K + seg * 16;
    const int ldsOff = rowL * 32 + seg * 16;

    // ---- prologue: stage tile 0 into buffer 0 ----
    stage_chunk<ABF>(Aptr, gA0, &sA[0][ldsOff]);
    stage_chunk<BBF>(Bptr, gB0, &sB[0][ldsOff]);
    wait_async_lds();
    __syncthreads();

    int cur = 0;
    for (int kt = 0; kt < K; kt += 32, cur ^= 1) {
        const bool hasNext = (kt + 32) < K;
        // ---- stage next tile into the other buffer (overlaps with WMMA) ----
        if (hasNext) {
            stage_chunk<ABF>(Aptr, gA0 + kt + 32, &sA[cur ^ 1][ldsOff]);
            stage_chunk<BBF>(Bptr, gB0 + kt + 32, &sB[cur ^ 1][ldsOff]);
        }

        // ---- fragments + WMMA from current buffer ----
        Frag16 a[2], b[4];
#pragma unroll
        for (int mt = 0; mt < 2; ++mt) {
            const uint4* p = (const uint4*)&sA[cur][(wm * 32 + mt * 16 + l16) * 32];
            a[mt].u[0] = p[half];
            a[mt].u[1] = p[2 + half];
        }
#pragma unroll
        for (int nt = 0; nt < 4; ++nt) {
            const uint4* p = (const uint4*)&sB[cur][(wn * 64 + nt * 16 + l16) * 32];
            b[nt].u[0] = p[half];
            b[nt].u[1] = p[2 + half];
        }
#pragma unroll
        for (int mt = 0; mt < 2; ++mt)
#pragma unroll
            for (int nt = 0; nt < 4; ++nt)
                acc[mt][nt] = __builtin_amdgcn_wmma_f32_16x16x32_bf16(
                    false, a[mt].v, false, b[nt].v, (short)0, acc[mt][nt],
                    false, false);

        if (hasNext) wait_async_lds();
        __syncthreads();
    }

    // ---- epilogue ----
#pragma unroll
    for (int mt = 0; mt < 2; ++mt) {
#pragma unroll
        for (int nt = 0; nt < 4; ++nt) {
#pragma unroll
            for (int e = 0; e < 8; ++e) {
                int row = blockM + wm * 32 + mt * 16 + e + 8 * half;
                int col = blockN + wn * 64 + nt * 16 + l16;
                float val = acc[mt][nt][e] * scale;
                if (MODE == 0) {
                    ((float*)Out)[(size_t)row * Nout + col] = val;
                } else if (MODE == 1) {
                    int b4 = row / NTOK, n = row % NTOK;
                    int h = col >> 5, r = col & 31;
                    ((unsigned short*)Out)[(((size_t)(b4 * HN + h) * NTOK + n) * RR) + r] =
                        f2bf(val);
                } else {
                    int b4 = row / NTOK, n = row % NTOK;
                    int h = col >> 7, d = col & 127;
                    ((unsigned short*)Out)[((size_t)(b4 * HN + h) * DKK + d) * NTOK + n] =
                        f2bf(val);
                }
            }
        }
    }
}

// ---------------------------------------------------------------------------
// Kernel 3: flash attention, one wave = 16 query rows, R=32 score contraction.
//   Scores computed TRANSPOSED: S^T = K_tile @ Q^T, so lane = query column and
//   each lane's 8 C-elements are keys -> softmax reductions are in-lane.
//   ALiBi bias split into wave-uniform regions:
//     A: all keys <= query  -> bias 0 (no VALU)
//     C: all keys >  query  -> affine bias (adds only, constants hoisted)
//     B: boundary (<=2 blocks) -> general clamped path
// ---------------------------------------------------------------------------
__global__ __launch_bounds__(256) void attn_flash_kernel(
    const unsigned short* __restrict__ qws,
    const unsigned short* __restrict__ kws,
    const unsigned short* __restrict__ vT,
    unsigned short* __restrict__ zws)
{
    __shared__ __align__(16) unsigned short sP[8 * 16 * 32];   // per-wave P tiles

    const int tid  = threadIdx.x;
    const int lane = tid & 31;
    const int wv   = tid >> 5;
    const int half = lane >> 4;
    const int l16  = lane & 15;

    const int h = blockIdx.y;
    const int b = blockIdx.z;
    const int qbase = blockIdx.x * 128 + wv * 16;
    const size_t bh = (size_t)(b * HN + h);
    const float slope = exp2f(-0.5f * (float)(h + 1));
    const float slope16 = 16.f * slope;
    const int iq = qbase + l16;               // this lane's query (stats owner)
    const float fiq = (float)iq;

    // hoisted per-lane bias constants: c0[e] = -slope*(e + 8*half)
    float c0[8];
#pragma unroll
    for (int e = 0; e < 8; ++e) c0[e] = -slope * (float)(e + 8 * half);

    // Q^T fragment (B-matrix 32x16: lane = query column; same bytes as A layout)
    Frag16 qf;
    {
        const uint4* p = (const uint4*)(qws + (bh * NTOK + iq) * RR);
        qf.u[0] = p[half];
        qf.u[1] = p[2 + half];
    }

    v8f o[8];
#pragma unroll
    for (int t = 0; t < 8; ++t)
#pragma unroll
        for (int e = 0; e < 8; ++e) o[t][e] = 0.f;

    float mrun = -1.0e30f;                    // running max for query iq
    float lrun = 0.f;                         // partial sum over THIS lane's keys

    unsigned short* myP = &sP[wv * 16 * 32];
    v8f zc;
#pragma unroll
    for (int e = 0; e < 8; ++e) zc[e] = 0.f;

    for (int kb = 0; kb < NTOK; kb += 32) {
        // K fragments (A-matrix 16x32: lane = key row, contiguous 64B rows)
        Frag16 kf0, kf1;
        {
            const uint4* p0 = (const uint4*)(kws + (bh * NTOK + kb + l16) * RR);
            kf0.u[0] = p0[half]; kf0.u[1] = p0[2 + half];
            const uint4* p1 = (const uint4*)(kws + (bh * NTOK + kb + 16 + l16) * RR);
            kf1.u[0] = p1[half]; kf1.u[1] = p1[2 + half];
        }
        // S^T tiles: C[key, query] -> lane holds one query, 8 keys per tile
        v8f s0 = __builtin_amdgcn_wmma_f32_16x16x32_bf16(false, kf0.v, false, qf.v,
                                                         (short)0, zc, false, false);
        v8f s1 = __builtin_amdgcn_wmma_f32_16x16x32_bf16(false, kf1.v, false, qf.v,
                                                         (short)0, zc, false, false);

        // ALiBi bias (key j vs query iq) - wave-uniform region split
        float x[16];
        if (kb + 31 <= qbase) {
            // Region A: every key <= every query in this wave -> bias = 0
#pragma unroll
            for (int e = 0; e < 8; ++e) {
                x[e]     = s0[e];
                x[8 + e] = s1[e];
            }
        } else if (kb >= qbase + 16) {
            // Region C: every key > every query -> pure affine bias
            float base = slope * (fiq - (float)kb);           // = -slope*(kb - iq)
#pragma unroll
            for (int e = 0; e < 8; ++e) {
                float b0 = base + c0[e];
                x[e]     = s0[e] + b0;
                x[8 + e] = s1[e] + b0 - slope16;
            }
        } else {
            // Region B: boundary blocks, general clamped path
#pragma unroll
            for (int e = 0; e < 8; ++e) {
                int j0 = kb + e + 8 * half;
                int j1 = j0 + 16;
                x[e]     = s0[e] - slope * (float)((j0 > iq) ? (j0 - iq) : 0);
                x[8 + e] = s1[e] - slope * (float)((j1 > iq) ? (j1 - iq) : 0);
            }
        }

        float tmax = x[0];
#pragma unroll
        for (int e = 1; e < 16; ++e) tmax = fmaxf(tmax, x[e]);
        // merge with partner half-lane (other 16 keys of this block)
        tmax = fmaxf(tmax, __shfl_xor(tmax, 16, 32));

        float mnew  = fmaxf(mrun, tmax);
        float alpha = __expf(mrun - mnew);
        mrun = mnew;

        float p[16];
        float ls = 0.f;
#pragma unroll
        for (int e = 0; e < 16; ++e) {
            p[e] = __expf(x[e] - mnew);
            ls += p[e];
        }
        lrun = lrun * alpha + ls;

        // store P^T -> P row-major [query][key]: lane's 8 values are consecutive
        {
            uint4* d0 = (uint4*)&myP[l16 * 32 + 8 * half];
            d0[0] = pack8f(&p[0]);
            uint4* d1 = (uint4*)&myP[l16 * 32 + 16 + 8 * half];
            d1[0] = pack8f(&p[8]);
        }

        // rescale O: alpha lives lane=query; O rows are queries e+8*half
        float arow[8];
#pragma unroll
        for (int e = 0; e < 8; ++e) arow[e] = __shfl(alpha, e + 8 * half, 32);
#pragma unroll
        for (int t2 = 0; t2 < 8; ++t2)
#pragma unroll
            for (int e = 0; e < 8; ++e) o[t2][e] *= arow[e];

        // wave-local LDS transpose round trip: wait on DS counter (CDNA5 split cnt)
        asm volatile("s_wait_dscnt 0" ::: "memory");

        Frag16 pa;
        {
            const uint4* p2 = (const uint4*)&myP[l16 * 32];
            pa.u[0] = p2[half];
            pa.u[1] = p2[2 + half];
        }
#pragma unroll
        for (int t2 = 0; t2 < 8; ++t2) {
            Frag16 vf;
            const uint4* p2 = (const uint4*)(vT + ((bh * DKK + t2 * 16 + l16) * (size_t)NTOK + kb));
            vf.u[0] = p2[half];
            vf.u[1] = p2[2 + half];
            o[t2] = __builtin_amdgcn_wmma_f32_16x16x32_bf16(false, pa.v, false, vf.v,
                                                            (short)0, o[t2], false, false);
        }
    }

    // final: merge partner partial sums, broadcast to row orientation, store z
    float ltot = lrun + __shfl_xor(lrun, 16, 32);
    float lrow[8];
#pragma unroll
    for (int e = 0; e < 8; ++e) lrow[e] = __shfl(ltot, e + 8 * half, 32);

#pragma unroll
    for (int t2 = 0; t2 < 8; ++t2) {
#pragma unroll
        for (int e = 0; e < 8; ++e) {
            int row = qbase + e + 8 * half;
            int d   = t2 * 16 + l16;
            float val = o[t2][e] / lrow[e];
            zws[((size_t)b * NTOK + row) * DD + h * DKK + d] = f2bf(val);
        }
    }
}

// ---------------------------------------------------------------------------
// Launch
// ---------------------------------------------------------------------------
extern "C" void kernel_launch(void* const* d_in, const int* in_sizes, int n_in,
                              void* d_out, int out_size, void* d_ws, size_t ws_size,
                              hipStream_t stream)
{
    const float* x     = (const float*)d_in[0];
    const float* Wq    = (const float*)d_in[1];
    const float* Wk    = (const float*)d_in[2];
    const float* Wv    = (const float*)d_in[3];
    const float* U     = (const float*)d_in[4];
    const float* Wproj = (const float*)d_in[5];
    (void)in_sizes; (void)n_in; (void)out_size; (void)ws_size;

    char* ws = (char*)d_ws;
    size_t off = 0;
    unsigned short* WqUt = (unsigned short*)(ws + off); off += (size_t)512 * DD * 2;
    unsigned short* WkUt = (unsigned short*)(ws + off); off += (size_t)512 * DD * 2;
    unsigned short* qws  = (unsigned short*)(ws + off); off += (size_t)BB * HN * NTOK * RR * 2;
    unsigned short* kws  = (unsigned short*)(ws + off); off += (size_t)BB * HN * NTOK * RR * 2;
    unsigned short* vT   = (unsigned short*)(ws + off); off += (size_t)BB * HN * DKK * NTOK * 2;
    unsigned short* zws  = (unsigned short*)(ws + off); off += (size_t)BB * NTOK * DD * 2;

    const int M = BB * NTOK;                    // 8192
    const float qscale = 0.08838834764831845f;  // 1/sqrt(128)

    // 1) effective low-rank weights (transposed, bf16)
    effw_kernel<<<dim3(DD / 256, 512, 2), 256, 0, stream>>>(Wq, Wk, U, WqUt, WkUt);

    // 2) q = x @ WqU (scaled), permuted to [B,H,N,R]
    gemm_wmma_kernel<false, true, 1><<<dim3(512 / 128, M / 128), 256, 0, stream>>>(
        x, WqUt, M, 512, DD, qws, qscale);

    // 3) k = x @ WkU, permuted to [B,H,N,R]
    gemm_wmma_kernel<false, true, 1><<<dim3(512 / 128, M / 128), 256, 0, stream>>>(
        x, WkUt, M, 512, DD, kws, 1.0f);

    // 4) v = x @ Wv^T, stored transposed [B,H,DK,N]
    gemm_wmma_kernel<false, false, 2><<<dim3(DD / 128, M / 128), 256, 0, stream>>>(
        x, Wv, M, DD, DD, vT, 1.0f);

    // 5) flash attention -> z [B,N,D] bf16
    attn_flash_kernel<<<dim3(NTOK / 128, HN, BB), 256, 0, stream>>>(qws, kws, vT, zws);

    // 6) out = z @ Wproj^T, fp32
    gemm_wmma_kernel<true, false, 0><<<dim3(DD / 128, M / 128), 256, 0, stream>>>(
        zws, Wproj, M, DD, DD, (float*)d_out, 1.0f);
}
